// GroupQueryAttention_88416196755613
// MI455X (gfx1250) — matrix operand order repaired and checked by reading, not compiled, not run on hardware
//
#include <hip/hip_runtime.h>
#include <hip/hip_bf16.h>

typedef __attribute__((ext_vector_type(16))) _Float16 v16h;
typedef __attribute__((ext_vector_type(8)))  float    v8f;
typedef __attribute__((ext_vector_type(4)))  unsigned int u32x4;

#define S_LEN   4096
#define D_MODEL 2048
#define NH      16
#define NG      4
#define HD      128
#define DKV     512   // NG*HD

// ---------------------------------------------------------------------------
// Fragment loaders (CDNA5 16-bit WMMA layouts, wave32).
// A (16x32, MxK): lane r=lane&15 holds row M=r; half=lane>>4;
//   VGPR0-3 hold K = half*8 + {0..7}, VGPR4-7 hold K = 16 + half*8 + {0..7}.
// B (32x16, KxN): lane holds column N=lane&15; same K order as A:
//   VGPR0-3 hold K = half*8 + {0..7}, VGPR4-7 hold K = 16 + half*8 + {0..7}.
// Both reduce to two 16B loads per lane from row-major (by M / by N) storage.
// ---------------------------------------------------------------------------
static __device__ __forceinline__ v16h ld_frag_a(const _Float16* p, int row,
                                                 int ldk, int kbase, int lane) {
    const int half = (lane >> 4) & 1;
    const _Float16* base = p + (size_t)row * ldk + kbase + half * 8;
    union { u32x4 u[2]; v16h v; } t;
    t.u[0] = *reinterpret_cast<const u32x4*>(base);
    t.u[1] = *reinterpret_cast<const u32x4*>(base + 16);
    return t.v;
}

static __device__ __forceinline__ v16h ld_frag_b(const _Float16* p, int row,
                                                 int ldk, int kbase, int lane) {
    const int half = (lane >> 4) & 1;
    const _Float16* base = p + (size_t)row * ldk + kbase + half * 8;
    union { u32x4 u[2]; v16h v; } t;
    t.u[0] = *reinterpret_cast<const u32x4*>(base);
    t.u[1] = *reinterpret_cast<const u32x4*>(base + 16);
    return t.v;
}

static __device__ __forceinline__ v8f wmma16(v16h a, v16h b, v8f c) {
    return __builtin_amdgcn_wmma_f32_16x16x32_f16(
        /*neg_a=*/false, a, /*neg_b=*/false, b,
        /*c_mod=*/(short)0, c, /*reuse_a=*/false, /*reuse_b=*/false);
}

// ---------------------------------------------------------------------------
// Elementwise helpers
// ---------------------------------------------------------------------------
__global__ void cvt_f16_kernel(const float* __restrict__ src,
                               _Float16* __restrict__ dst, int n) {
    int i = blockIdx.x * 256 + threadIdx.x;
    if (i < n) dst[i] = (_Float16)src[i];
}

// src is KxN fp32 row-major; dst is NxK f16 row-major (transposed)
__global__ void transpose_cvt_kernel(const float* __restrict__ src,
                                     _Float16* __restrict__ dst, int K, int N) {
    int i = blockIdx.x * 256 + threadIdx.x;
    if (i >= K * N) return;
    int k = i / N, n = i - k * N;
    dst[(size_t)n * K + k] = (_Float16)src[i];
}

// In-place RoPE on a (S, nheads*128) f16 matrix; one thread per (s,h,d<64) pair.
__global__ void rope_kernel(_Float16* __restrict__ T,
                            const float* __restrict__ cosp,
                            const float* __restrict__ sinp, int nheads) {
    int i = blockIdx.x * 256 + threadIdx.x;
    int total = S_LEN * nheads * 64;
    if (i >= total) return;
    int d = i & 63;
    int h = (i >> 6) % nheads;
    int s = i / (nheads * 64);
    _Float16* row = T + (size_t)s * (nheads * 128) + h * 128;
    float t1 = (float)row[d];
    float t2 = (float)row[d + 64];
    float c  = cosp[s * HD + d];   // cos[s][d] == cos[s][d+64] (concatenated)
    float sn = sinp[s * HD + d];
    row[d]      = (_Float16)(t1 * c - t2 * sn);
    row[d + 64] = (_Float16)(t2 * c + t1 * sn);
}

// ---------------------------------------------------------------------------
// WMMA GEMM: C(MxN) = A(MxK,f16 row-major) * Bt(NxK,f16 row-major)^T
// OUT_MODE: 0 = f16 row-major, 1 = f16 transposed (NxM), 2 = fp32 row-major
// Block: 256 threads = 8 waves; wave -> 64x32 output (8 WMMAs : 12 b128 loads
// per 32-K step, ~130 VGPRs so the compiler can K-unroll / double-buffer).
// Waves arranged 2(M) x 4(N) -> block tile 128x128.
// ---------------------------------------------------------------------------
template <int OUT_MODE>
__global__ __launch_bounds__(256)
void gemm_wmma_kernel(const _Float16* __restrict__ A,
                      const _Float16* __restrict__ Bt,
                      void* __restrict__ Cout, int M, int N, int K) {
    const int lane  = threadIdx.x & 31;
    const int wave  = threadIdx.x >> 5;
    const int mbase = blockIdx.y * 128 + (wave & 1) * 64;
    const int nbase = blockIdx.x * 128 + (wave >> 1) * 32;
    const int r     = lane & 15;

    v8f acc[4][2] = {};
    for (int kb = 0; kb < K; kb += 32) {
        if (kb + 32 < K) {  // speculative prefetch of next K slab
            __builtin_prefetch(A + (size_t)(mbase + r) * K + kb + 32, 0, 1);
            __builtin_prefetch(Bt + (size_t)(nbase + r) * K + kb + 32, 0, 1);
        }
        v16h a[4], b[2];
#pragma unroll
        for (int i = 0; i < 4; ++i)
            a[i] = ld_frag_a(A, mbase + i * 16 + r, K, kb, lane);
#pragma unroll
        for (int i = 0; i < 2; ++i)
            b[i] = ld_frag_b(Bt, nbase + i * 16 + r, K, kb, lane);
#pragma unroll
        for (int mi = 0; mi < 4; ++mi)
#pragma unroll
            for (int ni = 0; ni < 2; ++ni)
                acc[mi][ni] = wmma16(a[mi], b[ni], acc[mi][ni]);
    }

    const int half = (lane >> 4) & 1;
#pragma unroll
    for (int mi = 0; mi < 4; ++mi)
#pragma unroll
        for (int ni = 0; ni < 2; ++ni)
#pragma unroll
            for (int j = 0; j < 8; ++j) {
                int m = mbase + mi * 16 + half * 8 + j;
                int n = nbase + ni * 16 + r;
                float v = acc[mi][ni][j];
                if (OUT_MODE == 2)
                    ((float*)Cout)[(size_t)m * N + n] = v;
                else if (OUT_MODE == 1)
                    ((_Float16*)Cout)[(size_t)n * M + m] = (_Float16)v;
                else
                    ((_Float16*)Cout)[(size_t)m * N + n] = (_Float16)v;
            }
}

// ---------------------------------------------------------------------------
// Flash attention: one wave handles (head, 16 query rows); key tiles of 32.
// Qh: (S, 2048) f16 (rope'd); Kh: (S, 512) f16 (rope'd); Vt: (512, S) f16.
// Ch: (S, 2048) f16 context output.  (No prefetch: K/V are L2-resident.)
// ---------------------------------------------------------------------------
__global__ __launch_bounds__(128)
void attn_kernel(const _Float16* __restrict__ Qh, const _Float16* __restrict__ Kh,
                 const _Float16* __restrict__ Vt, _Float16* __restrict__ Ch) {
    __shared__ __align__(16) _Float16 Plds[4][16][32];

    const int lane  = threadIdx.x & 31;
    const int wave  = threadIdx.x >> 5;
    const int head  = blockIdx.x >> 6;        // 16 heads
    const int chunk = blockIdx.x & 63;        // 64 chunks of 64 q-rows
    const int g     = head >> 2;              // KV group
    const int qbase = chunk * 64 + wave * 16;
    const int r     = lane & 15;
    const int half  = (lane >> 4) & 1;
    const float scale = 0.08838834764831845f;  // 1/sqrt(128)
    const float LOG2E = 1.4426950408889634f;

    // Q fragments for this wave's 16 rows, K-dim = 128 -> 4 frags
    v16h qf[4];
#pragma unroll
    for (int ck = 0; ck < 4; ++ck)
        qf[ck] = ld_frag_a(Qh, qbase + r, D_MODEL, head * HD + ck * 32, lane);

    float m[8], l[8];
    v8f acc[8] = {};
#pragma unroll
    for (int j = 0; j < 8; ++j) { m[j] = -3.0e38f; l[j] = 0.0f; }

    for (int kt = 0; kt < qbase + 16; kt += 32) {
        // scores: two 16x16 tiles over 32 keys, K-dim 128 = 4 WMMA steps each
        v8f c0 = {}, c1 = {};
#pragma unroll
        for (int ck = 0; ck < 4; ++ck) {
            v16h k0 = ld_frag_b(Kh, kt + r,      DKV, g * HD + ck * 32, lane);
            v16h k1 = ld_frag_b(Kh, kt + 16 + r, DKV, g * HD + ck * 32, lane);
            c0 = wmma16(qf[ck], k0, c0);
            c1 = wmma16(qf[ck], k1, c1);
        }

#pragma unroll
        for (int j = 0; j < 8; ++j) {
            int qrow = qbase + half * 8 + j;
            float s0 = c0[j] * scale;
            float s1 = c1[j] * scale;
            if (kt + r > qrow)      s0 = -3.0e38f;   // causal mask
            if (kt + 16 + r > qrow) s1 = -3.0e38f;
            // row max over the 16 key-columns (lanes within each 16-half)
            float rm = fmaxf(s0, s1);
            rm = fmaxf(rm, __shfl_xor(rm, 1));
            rm = fmaxf(rm, __shfl_xor(rm, 2));
            rm = fmaxf(rm, __shfl_xor(rm, 4));
            rm = fmaxf(rm, __shfl_xor(rm, 8));
            float mn    = fmaxf(m[j], rm);
            float alpha = exp2f((m[j] - mn) * LOG2E);
            float p0    = exp2f((s0 - mn) * LOG2E);
            float p1    = exp2f((s1 - mn) * LOG2E);
            float rs = p0 + p1;
            rs += __shfl_xor(rs, 1);
            rs += __shfl_xor(rs, 2);
            rs += __shfl_xor(rs, 4);
            rs += __shfl_xor(rs, 8);
            l[j] = l[j] * alpha + rs;
            m[j] = mn;
            // stage P (C layout -> LDS) for A-layout reload
            Plds[wave][half * 8 + j][r]      = (_Float16)p0;
            Plds[wave][half * 8 + j][16 + r] = (_Float16)p1;
#pragma unroll
            for (int nt = 0; nt < 8; ++nt) acc[nt][j] *= alpha;
        }

        v16h pf = ld_frag_a(&Plds[wave][0][0], r, 32, 0, lane);
#pragma unroll
        for (int nt = 0; nt < 8; ++nt) {
            v16h vf = ld_frag_b(Vt, g * HD + nt * 16 + r, S_LEN, kt, lane);
            acc[nt] = wmma16(pf, vf, acc[nt]);
        }
    }

#pragma unroll
    for (int j = 0; j < 8; ++j) {
        float inv = 1.0f / l[j];
        int qrow = qbase + half * 8 + j;
#pragma unroll
        for (int nt = 0; nt < 8; ++nt)
            Ch[(size_t)qrow * D_MODEL + head * HD + nt * 16 + r] =
                (_Float16)(acc[nt][j] * inv);
    }
}

// ---------------------------------------------------------------------------
// Host-side launcher
// ---------------------------------------------------------------------------
extern "C" void kernel_launch(void* const* d_in, const int* in_sizes, int n_in,
                              void* d_out, int out_size, void* d_ws, size_t ws_size,
                              hipStream_t stream) {
    const float* x    = (const float*)d_in[0];
    // d_in[1] = mask (unused; causal mask computed in-kernel)
    const float* wq   = (const float*)d_in[2];
    const float* wk   = (const float*)d_in[3];
    const float* wv   = (const float*)d_in[4];
    const float* wo   = (const float*)d_in[5];
    const float* cosp = (const float*)d_in[6];
    const float* sinp = (const float*)d_in[7];

    _Float16* p   = (_Float16*)d_ws;
    _Float16* xh  = p; p += (size_t)S_LEN * D_MODEL;
    _Float16* wqh = p; p += (size_t)D_MODEL * D_MODEL;   // transposed NxK
    _Float16* wkh = p; p += (size_t)D_MODEL * DKV;
    _Float16* wvh = p; p += (size_t)D_MODEL * DKV;
    _Float16* woh = p; p += (size_t)D_MODEL * D_MODEL;
    _Float16* Qh  = p; p += (size_t)S_LEN * D_MODEL;
    _Float16* Kh  = p; p += (size_t)S_LEN * DKV;
    _Float16* Vt  = p; p += (size_t)S_LEN * DKV;         // transposed (512 x S)
    _Float16* Ch  = p; p += (size_t)S_LEN * D_MODEL;

    int n;
    n = S_LEN * D_MODEL;
    cvt_f16_kernel<<<(n + 255) / 256, 256, 0, stream>>>(x, xh, n);
    n = D_MODEL * D_MODEL;
    transpose_cvt_kernel<<<(n + 255) / 256, 256, 0, stream>>>(wq, wqh, D_MODEL, D_MODEL);
    n = D_MODEL * DKV;
    transpose_cvt_kernel<<<(n + 255) / 256, 256, 0, stream>>>(wk, wkh, D_MODEL, DKV);
    transpose_cvt_kernel<<<(n + 255) / 256, 256, 0, stream>>>(wv, wvh, D_MODEL, DKV);
    n = D_MODEL * D_MODEL;
    transpose_cvt_kernel<<<(n + 255) / 256, 256, 0, stream>>>(wo, woh, D_MODEL, D_MODEL);

    gemm_wmma_kernel<0><<<dim3(D_MODEL / 128, S_LEN / 128), 256, 0, stream>>>(
        xh, wqh, Qh, S_LEN, D_MODEL, D_MODEL);
    gemm_wmma_kernel<0><<<dim3(DKV / 128, S_LEN / 128), 256, 0, stream>>>(
        xh, wkh, Kh, S_LEN, DKV, D_MODEL);
    gemm_wmma_kernel<1><<<dim3(DKV / 128, S_LEN / 128), 256, 0, stream>>>(
        xh, wvh, Vt, S_LEN, DKV, D_MODEL);

    n = S_LEN * NH * 64;
    rope_kernel<<<(n + 255) / 256, 256, 0, stream>>>(Qh, cosp, sinp, NH);
    n = S_LEN * NG * 64;
    rope_kernel<<<(n + 255) / 256, 256, 0, stream>>>(Kh, cosp, sinp, NG);

    attn_kernel<<<NH * (S_LEN / 64), 128, 0, stream>>>(Qh, Kh, Vt, Ch);

    gemm_wmma_kernel<2><<<dim3(D_MODEL / 128, S_LEN / 128), 256, 0, stream>>>(
        Ch, woh, d_out, S_LEN, D_MODEL, D_MODEL);
}
